// WeightOnlyQuantizedLinearImpl_27848567947269
// MI455X (gfx1250) — matrix operand order repaired and checked by reading, not compiled, not run on hardware
//
#include <hip/hip_runtime.h>

// ---------------------------------------------------------------------------
// w8a16 GEMM for MI455X (gfx1250):
//   out[M,N] = A[M,K](f32) @ dequant_fp8_e4m3fn(Q[K,N]) * scales[K/128,N] + bias
//
// Roofline: 369 GFLOP, ~424 MB unique traffic -> compute-bound; use
// V_WMMA_F32_16X16X32_F16 (f16 frags, f32 accum; fp8-WMMA would break the
// reference's NaN->0 rule and fp8 activations cost ~2% error).
//
// Scheduling: grid is (M-fast, N): consecutive WGs share one 2MB Q band
// (L2-hit) and stream A, which fits in the 192MB L2 and stays resident ->
// HBM traffic ~= Q(180MB)+A(64MB)+out(180MB) ~ 18us @ 23.3TB/s.
//
// Dequant: packed f16-domain decode (~3 ops/weight): v_perm byte-pack ->
// shift/add exponent rebias -> NaN mask (v_pk_sub_i16 trick) -> sign OR ->
// v_pk_mul_f16 by the group scale. BM=256 halves per-output dequant work.
// ---------------------------------------------------------------------------

typedef __attribute__((ext_vector_type(16))) _Float16 v16h;
typedef __attribute__((ext_vector_type(8)))  float    v8f;
typedef __attribute__((ext_vector_type(2)))  _Float16 h2;
typedef __attribute__((ext_vector_type(2)))  unsigned short u16x2;

#define K_DIM   4096
#define N_DIM   11008
#define M_DIM   4096
#define BM      256
#define BN      128
#define BK      64
#define NTHR    512                  // 16 waves
#define KSTEPS  (K_DIM / BK)         // 64

// LDS geometry (in dwords)
#define A_ROW_DW   36                       // 64 halves + 8 pad halves = 144B
#define A_LDS_DW   (256 * A_ROW_DW)         // 36 KB
#define B_LANE_DW  12                       // 48B lane slot: conflict-free b128
#define B_F_DW     (32 * B_LANE_DW)         // 384
#define B_KH_DW    (8 * B_F_DW)             // 3072
#define B_LDS_DW   (2 * B_KH_DW)            // 24 KB

// Decode two fp8 e4m3fn bytes (k, k+1 of one column) + scale -> packed f16.
// w0/w1 are int32 with the byte in [7:0] and zero above (qweight values 0..255).
__device__ __forceinline__ unsigned dec_pair(int w0, int w1, h2 sc) {
    // bytes: [0]=w0.b0, [1]=w0.b1(=0), [2]=w1.b0, [3]=w1.b1(=0)
    unsigned p  = __builtin_amdgcn_perm((unsigned)w1, (unsigned)w0, 0x05040100u);
    unsigned l7 = p & 0x007F007Fu;                    // {e4,m3} per 16-bit lane
    unsigned t  = (l7 << 7) + 0x20002000u;            // f16 magnitude (exp += 8)
    // NaN (l7 == 0x7F) -> zero that half
    unsigned nb = ((l7 + 0x00010001u) & 0x00800080u) >> 7;  // 1 per NaN half
    u16x2 z = {0, 0};
    u16x2 nz = z - __builtin_bit_cast(u16x2, nb);     // 0xFFFF per NaN half
    t &= ~__builtin_bit_cast(unsigned, nz);
    t |= (p & 0x00800080u) << 8;                      // sign bits
    h2 v = __builtin_bit_cast(h2, t) * sc;            // v_pk_mul_f16
    return __builtin_bit_cast(unsigned, v);
}

extern "C" __global__ __launch_bounds__(NTHR)
void wq_gemm_f16wmma(const float* __restrict__ A, const int* __restrict__ Q,
                     const float* __restrict__ S, const float* __restrict__ bias,
                     float* __restrict__ out)
{
    __shared__ __align__(16) unsigned aLds[A_LDS_DW];
    __shared__ __align__(16) unsigned bLds[B_LDS_DW];

    const int t    = threadIdx.x;
    const int lane = t & 31;
    const int wid  = t >> 5;            // 0..15
    const int wr   = wid >> 2;          // 0..3 : 64 M-rows each
    const int wc   = wid & 3;           // 0..3 : 32 N-cols each

    const int m0 = blockIdx.x * BM;     // M is the FAST grid dim (L2 reuse)
    const int n0 = blockIdx.y * BN;

    // A staging: 8x float4 per thread (rows i*32 + t/16, float4-col t%16)
    const int a_row32 = t >> 4;         // 0..31
    const int a_kq    = t & 15;
    // B staging: thread owns n-quad q (4 cols) over 4 k-rows
    const int q       = t & 31;
    const int kbase   = (t >> 5) * 4;   // 0..60
    const int bf      = q >> 2;
    const int nlo     = (q & 3) * 4;
    const int vbase   = (kbase & 15) >> 1;
    const int b_base_dw = (kbase >> 5) * B_KH_DW + bf * B_F_DW
                        + (nlo + (kbase & 16)) * B_LANE_DW;

    float4 as4[8];
    int4   bs4[4];
    float4 sc4;

    auto load_stage = [&](int s) {
        const int k0 = s * BK;
        const float4* A4 = (const float4*)A;
#pragma unroll
        for (int i = 0; i < 8; ++i) {
            int m = m0 + i * 32 + a_row32;
            as4[i] = A4[m * (K_DIM / 4) + (k0 >> 2) + a_kq];
        }
        const int4* Q4 = (const int4*)Q;
#pragma unroll
        for (int j = 0; j < 4; ++j) {
            bs4[j] = Q4[(k0 + kbase + j) * (N_DIM / 4) + (n0 >> 2) + q];
        }
        sc4 = *(const float4*)&S[(k0 >> 7) * N_DIM + n0 + q * 4];
    };

    auto store_stage = [&]() {
        // A: f32 -> packed f16, row-major [m][72 halves]
#pragma unroll
        for (int i = 0; i < 8; ++i) {
            int d = (i * 32 + a_row32) * A_ROW_DW + a_kq * 2;
            aLds[d]     = __builtin_bit_cast(unsigned,
                              __builtin_amdgcn_cvt_pkrtz(as4[i].x, as4[i].y));
            aLds[d + 1] = __builtin_bit_cast(unsigned,
                              __builtin_amdgcn_cvt_pkrtz(as4[i].z, as4[i].w));
        }
        // B: packed decode+scale, store pre-swizzled in B-fragment layout
        h2 sch[4];
        sch[0] = h2{(_Float16)sc4.x, (_Float16)sc4.x};
        sch[1] = h2{(_Float16)sc4.y, (_Float16)sc4.y};
        sch[2] = h2{(_Float16)sc4.z, (_Float16)sc4.z};
        sch[3] = h2{(_Float16)sc4.w, (_Float16)sc4.w};
#pragma unroll
        for (int jp = 0; jp < 2; ++jp) {
            const int4 wa = bs4[2 * jp];
            const int4 wb = bs4[2 * jp + 1];
            const int wa_[4] = {wa.x, wa.y, wa.z, wa.w};
            const int wb_[4] = {wb.x, wb.y, wb.z, wb.w};
            const int vv = vbase + jp;
#pragma unroll
            for (int jj = 0; jj < 4; ++jj) {
                bLds[b_base_dw + jj * B_LANE_DW + vv] =
                    dec_pair(wa_[jj], wb_[jj], sch[jj]);
            }
        }
    };

    v8f acc[4][2];
#pragma unroll
    for (int i = 0; i < 4; ++i)
#pragma unroll
        for (int j = 0; j < 2; ++j) acc[i][j] = (v8f){0, 0, 0, 0, 0, 0, 0, 0};

    const int a_rd_row = wr * 64 + (lane & 15);
    const int a_rd_off = (lane >> 4) * 8;        // dwords (32B k-phase offset)
    const int b_rd     = lane * B_LANE_DW;

    load_stage(0);
    store_stage();

    for (int s = 0; s < KSTEPS; ++s) {
        __syncthreads();                         // staged tile visible
        if (s + 1 < KSTEPS) load_stage(s + 1);   // hide global latency under WMMA

#pragma unroll
        for (int kh = 0; kh < 2; ++kh) {
            union { uint4 u[2]; v16h h; } fb[2];
#pragma unroll
            for (int ns = 0; ns < 2; ++ns) {
                int d = kh * B_KH_DW + (wc * 2 + ns) * B_F_DW + b_rd;
                fb[ns].u[0] = *(const uint4*)&bLds[d];
                fb[ns].u[1] = *(const uint4*)&bLds[d + 4];
            }
#pragma unroll
            for (int ms = 0; ms < 4; ++ms) {
                union { uint4 u[2]; v16h h; } fa;
                int d = (a_rd_row + ms * 16) * A_ROW_DW + kh * 16 + a_rd_off;
                fa.u[0] = *(const uint4*)&aLds[d];
                fa.u[1] = *(const uint4*)&aLds[d + 4];
#pragma unroll
                for (int ns = 0; ns < 2; ++ns) {
                    acc[ms][ns] = __builtin_amdgcn_wmma_f32_16x16x32_f16(
                        false, fa.h, false, fb[ns].h, (short)0, acc[ms][ns],
                        false, false);
                }
            }
        }
        __syncthreads();                         // all fragment reads done
        if (s + 1 < KSTEPS) store_stage();
    }

    // Epilogue: C layout — VGPR r: M = r + 8*(lane>=16), N = lane%16
    const int ncol_base = n0 + wc * 32 + (lane & 15);
    const int mrow_base = m0 + wr * 64 + (lane >> 4) * 8;
#pragma unroll
    for (int ns = 0; ns < 2; ++ns) {
        const int ncol = ncol_base + ns * 16;
        const float bv = bias[ncol];
#pragma unroll
        for (int ms = 0; ms < 4; ++ms) {
            const int mr = mrow_base + ms * 16;
#pragma unroll
            for (int r = 0; r < 8; ++r) {
                out[(mr + r) * N_DIM + ncol] = acc[ms][ns][r] + bv;
            }
        }
    }
}

extern "C" void kernel_launch(void* const* d_in, const int* in_sizes, int n_in,
                              void* d_out, int out_size, void* d_ws, size_t ws_size,
                              hipStream_t stream) {
    const float* input  = (const float*)d_in[0];
    const int*   qw     = (const int*)d_in[1];
    const float* scales = (const float*)d_in[2];
    const float* bias   = (const float*)d_in[3];
    float*       out    = (float*)d_out;

    dim3 grid(M_DIM / BM, N_DIM / BN);   // 16 x 86, M fastest for L2 reuse
    wq_gemm_f16wmma<<<grid, dim3(NTHR), 0, stream>>>(input, qw, scales, bias, out);

    (void)in_sizes; (void)n_in; (void)out_size; (void)d_ws; (void)ws_size;
}
